// T3ROLAND_6442450944064
// MI455X (gfx1250) — matrix operand order repaired and checked by reading, not compiled, not run on hardware
//
#include <hip/hip_runtime.h>

// ---------------------------------------------------------------------------
// MI455X (gfx1250) 2-layer temporal GCN.
//  - All dense GEMMs run on v_wmma_f32_16x16x32_bf16 with bf16-native operands
//    (converted once per launch, weights pre-transposed to [N,K]) so the hot
//    loop is pure uint4 LDS staging + ds_load_b128 fragments + WMMA.
//  - Edge aggregation is the HBM-bound part: coalesced float4 gathers +
//    global f32 atomics.
// ---------------------------------------------------------------------------

typedef __attribute__((ext_vector_type(16))) __bf16 bf16x16;
typedef __attribute__((ext_vector_type(8)))  float  f32x8;
typedef unsigned short u16b;

__device__ __forceinline__ u16b f2bf(float f) {
  // round-to-nearest-even fp32 -> bf16
  unsigned int u = __float_as_uint(f);
  u += 0x7FFFu + ((u >> 16) & 1u);
  return (u16b)(u >> 16);
}

#define BM 128
#define BN 64
#define BK 32

// C = act(A[M,K](bf16) @ WT[N,K](bf16)^T + bias (+ Cf if accum)).
// Outputs: Cf (f32, nullable) and/or Cb (bf16, nullable), both [M,N].
// Block: 128 threads = 4 waves; wave w owns rows [w*32, w*32+32) of the
// 128x64 block tile -> 8 WMMAs per K-step, B-fragments reused across the
// two row subtiles.
__global__ __launch_bounds__(128)
void k_gemm_bf(const u16b* __restrict__ A, int lda,
               const u16b* __restrict__ WT, int ldwt,
               const float* __restrict__ bias,
               float* __restrict__ Cf, u16b* __restrict__ Cb, int ldc,
               int M, int N, int K, int relu, int accum)
{
  __shared__ __align__(16) u16b As[BM * BK];   // 128x32 bf16 (8 KB)
  __shared__ __align__(16) u16b Ws[BN * BK];   // 64x32 bf16, [n][k] (4 KB)

  const int tid  = threadIdx.x;
  const int lane = tid & 31;
  const int wave = tid >> 5;
  const int rowBase = blockIdx.x * BM;
  const int colBase = blockIdx.y * BN;
  const int ml = lane & 15;
  const int hk = lane >> 4;

  f32x8 acc[2][4] = {};   // 2 row-subtiles x 4 col-subtiles (16x16 f32 each)

  const int nK = K / BK;
  for (int kt = 0; kt < nK; ++kt) {
    const int kBase = kt * BK;

    // Stage A: 128 rows x 4 uint4/row = 512 uint4; 4 per thread. Pure copy.
    #pragma unroll
    for (int i = 0; i < 4; ++i) {
      int p = tid + i * 128;
      int r = p >> 2;
      int q = p & 3;
      int grow = rowBase + r;
      uint4 v = make_uint4(0u, 0u, 0u, 0u);
      if (grow < M) v = *(const uint4*)(A + (size_t)grow * lda + kBase + q * 8);
      *(uint4*)&As[r * BK + q * 8] = v;
    }
    // Stage WT rows: 64 rows x 4 uint4 = 256 uint4; 2 per thread.
    #pragma unroll
    for (int i = 0; i < 2; ++i) {
      int p = tid + i * 128;
      int r = p >> 2;
      int q = p & 3;
      *(uint4*)&Ws[r * BK + q * 8] =
          *(const uint4*)(WT + (size_t)(colBase + r) * ldwt + kBase + q * 8);
    }
    // Prefetch next K-chunk of the weight panel (global_prefetch_b8).
    if (kt + 1 < nK)
      __builtin_prefetch(WT + (size_t)(colBase + (tid & 63)) * ldwt + kBase + BK, 0, 1);
    __syncthreads();

    // A fragments (16x32 bf16 each): lane = (m = lane&15, hk = lane>>4);
    // per-lane halves live at k = hk*8 and k = 16 + hk*8 -> two ds_load_b128.
    union F { uint4 q[2]; bf16x16 v; };
    F a0, a1;
    const int m0 = wave * 32 + ml;
    a0.q[0] = *(const uint4*)&As[m0 * BK + hk * 8];
    a0.q[1] = *(const uint4*)&As[m0 * BK + 16 + hk * 8];
    const int m1 = m0 + 16;
    a1.q[0] = *(const uint4*)&As[m1 * BK + hk * 8];
    a1.q[1] = *(const uint4*)&As[m1 * BK + 16 + hk * 8];

    #pragma unroll
    for (int nb = 0; nb < 4; ++nb) {
      // B fragment (32x16): lane holds k = hk*8 .. hk*8+7 and 16+hk*8 .. 16+hk*8+7 of column n.
      F b;
      const int n = nb * 16 + ml;
      b.q[0] = *(const uint4*)&Ws[n * BK + hk * 8];
      b.q[1] = *(const uint4*)&Ws[n * BK + 16 + hk * 8];
      acc[0][nb] = __builtin_amdgcn_wmma_f32_16x16x32_bf16(
          false, a0.v, false, b.v, (short)0, acc[0][nb], false, false);
      acc[1][nb] = __builtin_amdgcn_wmma_f32_16x16x32_bf16(
          false, a1.v, false, b.v, (short)0, acc[1][nb], false, false);
    }
    __syncthreads();
  }

  // Epilogue. C/D layout: col = lane&15; VGPR v -> row v + (lane>>4)*8.
  #pragma unroll
  for (int sub = 0; sub < 2; ++sub) {
    #pragma unroll
    for (int nb = 0; nb < 4; ++nb) {
      int col = colBase + nb * 16 + ml;
      float b = bias ? bias[col] : 0.0f;
      #pragma unroll
      for (int v = 0; v < 8; ++v) {
        int row = rowBase + wave * 32 + sub * 16 + hk * 8 + v;
        if (row < M) {
          float r = acc[sub][nb][v] + b;
          size_t off = (size_t)row * ldc + col;
          if (accum) r += Cf[off];
          if (relu)  r = fmaxf(r, 0.0f);
          if (Cf) Cf[off] = r;
          if (Cb) Cb[off] = f2bf(r);
        }
      }
    }
  }
}

// ---------------------------------------------------------------------------

__global__ void k_fill(float* __restrict__ p, long long n, float v) {
  long long i = (long long)blockIdx.x * blockDim.x + threadIdx.x;
  if (i < n) p[i] = v;
}

// fp32 -> bf16, 2 elements per thread (packed u32 store). n must be even.
__global__ void k_f2b(const float* __restrict__ in, u16b* __restrict__ out, long long n2) {
  long long i = (long long)blockIdx.x * blockDim.x + threadIdx.x;
  if (i >= n2) return;
  float2 v = *(const float2*)(in + 2 * i);
  unsigned int pk = (unsigned int)f2bf(v.x) | ((unsigned int)f2bf(v.y) << 16);
  *(unsigned int*)(out + 2 * i) = pk;
}

// WT[n*K + k] = bf16(W[k*N + n])  (weight convert + transpose, tiny)
__global__ void k_wt(const float* __restrict__ W, u16b* __restrict__ WT, int K, int N) {
  int idx = blockIdx.x * blockDim.x + threadIdx.x;
  if (idx >= N * K) return;
  int n = idx / K, k = idx - n * K;
  WT[idx] = f2bf(W[(size_t)k * N + n]);
}

__global__ void k_deg(const int* __restrict__ dst, float* __restrict__ deg, int E) {
  int e = blockIdx.x * blockDim.x + threadIdx.x;
  if (e < E) atomicAdd(&deg[dst[e]], 1.0f);
}

__global__ void k_dinv(const float* __restrict__ deg, float* __restrict__ dinv, int n) {
  int i = blockIdx.x * blockDim.x + threadIdx.x;
  if (i < n) dinv[i] = rsqrtf(deg[i] + 1.0f);  // +1 = self loop
}

// agg[dst] += hw[src] * dinv[src]*dinv[dst]; thread = (edge, 4 features)
__global__ void k_scatter(const float* __restrict__ hw, const float* __restrict__ dinv,
                          const int* __restrict__ src, const int* __restrict__ dst,
                          float* __restrict__ agg, long long total, int H, int cshift)
{
  long long idx = (long long)blockIdx.x * blockDim.x + threadIdx.x;
  if (idx >= total) return;
  int e = (int)(idx >> cshift);
  int c = ((int)idx & ((1 << cshift) - 1)) << 2;
  int s = src[e], d = dst[e];
  float norm = dinv[s] * dinv[d];
  float4 v = *(const float4*)(hw + (size_t)s * H + c);
  float* out = agg + (size_t)d * H + c;
  atomicAdd(out + 0, v.x * norm);
  atomicAdd(out + 1, v.y * norm);
  atomicAdd(out + 2, v.z * norm);
  atomicAdd(out + 3, v.w * norm);
}

// out_bf[i][c] = bf16(relu(agg[i][c] + hw[i][c]*dinv[i]^2 + bias[c]))
__global__ void k_gcn_epilogue(const float* __restrict__ agg, const float* __restrict__ hw,
                               const float* __restrict__ dinv, const float* __restrict__ bias,
                               u16b* __restrict__ out_bf, long long total, int H, int hshift)
{
  long long idx = (long long)blockIdx.x * blockDim.x + threadIdx.x;
  if (idx >= total) return;
  int i = (int)(idx >> hshift);
  int c = (int)idx & (H - 1);
  float di = dinv[i];
  float r = agg[idx] + hw[idx] * di * di + bias[c];
  out_bf[idx] = f2bf(fmaxf(r, 0.0f));
}

// pred[i] = sum_j emb1[a][j]*emb1[b][j]*(Wpost[j,0]+Wpost[j,1]) + bpost[0]+bpost[1]
__global__ void k_decode(const float* __restrict__ emb1,
                         const int* __restrict__ ea, const int* __restrict__ eb,
                         const float* __restrict__ Wpost, const float* __restrict__ bpost,
                         float* __restrict__ pred, int L, int H)
{
  int i = blockIdx.x * blockDim.x + threadIdx.x;
  if (i >= L) return;
  const float* pa = emb1 + (size_t)ea[i] * H;
  const float* pb = emb1 + (size_t)eb[i] * H;
  float s = 0.0f;
  for (int j = 0; j < H; j += 4) {
    float4 a = *(const float4*)(pa + j);
    float4 b = *(const float4*)(pb + j);
    s += a.x * b.x * (Wpost[2 * (j + 0)] + Wpost[2 * (j + 0) + 1]);
    s += a.y * b.y * (Wpost[2 * (j + 1)] + Wpost[2 * (j + 1) + 1]);
    s += a.z * b.z * (Wpost[2 * (j + 2)] + Wpost[2 * (j + 2) + 1]);
    s += a.w * b.w * (Wpost[2 * (j + 3)] + Wpost[2 * (j + 3) + 1]);
  }
  pred[i] = s + bpost[0] + bpost[1];
}

// ---------------------------------------------------------------------------

static inline int cdiv(long long a, long long b) { return (int)((a + b - 1) / b); }

extern "C" void kernel_launch(void* const* d_in, const int* in_sizes, int n_in,
                              void* d_out, int out_size, void* d_ws, size_t ws_size,
                              hipStream_t stream) {
  const float* x     = (const float*)d_in[0];
  const float* prev0 = (const float*)d_in[1];
  const float* prev1 = (const float*)d_in[2];
  const float* Wpre1 = (const float*)d_in[3];
  const float* bpre1 = (const float*)d_in[4];
  const float* Wpre2 = (const float*)d_in[5];
  const float* bpre2 = (const float*)d_in[6];
  const float* Wc1   = (const float*)d_in[7];
  const float* bc1   = (const float*)d_in[8];
  const float* Wc2   = (const float*)d_in[9];
  const float* bc2   = (const float*)d_in[10];
  const float* Wm1   = (const float*)d_in[11];
  const float* bm1   = (const float*)d_in[12];
  const float* Wm2   = (const float*)d_in[13];
  const float* bm2   = (const float*)d_in[14];
  const float* Wpost = (const float*)d_in[15];
  const float* bpost = (const float*)d_in[16];
  const int*   ei    = (const int*)d_in[17];   // [2, E]
  const int*   eli   = (const int*)d_in[18];   // [2, L]

  const int DIN = 256, H1 = 128, H2 = 64;
  const int N = in_sizes[0] / DIN;             // 50000 nodes
  const int E = in_sizes[17] / 2;              // 1.6M edges
  const int L = in_sizes[18] / 2;              // 200k label edges

  // ----- workspace layout -----
  // f32 region: deg | dinv | hw1[N,128] | agg1[N,128] | hw2[N,64] | agg2[N,64]
  float* ws = (float*)d_ws;
  size_t nA = ((size_t)N + 127) & ~(size_t)127;
  float* deg  = ws;
  float* dinv = ws + nA;
  float* hw1  = ws + 2 * nA;
  float* agg1 = hw1 + (size_t)N * H1;
  float* hw2  = agg1 + (size_t)N * H1;
  float* agg2 = hw2 + (size_t)N * H2;
  // bf16 region (aliased by lifetime):
  //  bfA[N*256]: x_bf            -> conv1_bf (N*128) + emb0_bf (N*128)
  //  bfB[N*256]: h1_bf           -> prev0_bf (N*128) + prev1_bf (N*64)
  //  bfC[N*128]: h2_bf           -> conv2_bf (N*64)
  u16b* bf0 = (u16b*)(agg2 + (size_t)N * H2);
  u16b* wt  = bf0;                               // 163840 bf16 weights
  u16b* WTpre1 = wt;                             // 256x256
  u16b* WTpre2 = WTpre1 + 256 * 256;             // 128x256
  u16b* WTc1   = WTpre2 + 128 * 256;             // 128x128
  u16b* WTm1a  = WTc1   + 128 * 128;             // 128x128
  u16b* WTm1b  = WTm1a  + 128 * 128;             // 128x128
  u16b* WTc2   = WTm1b  + 128 * 128;             // 64x128
  u16b* WTm2a  = WTc2   + 64 * 128;              // 64x64
  u16b* WTm2b  = WTm2a  + 64 * 64;               // 64x64
  u16b* bfA = WTm2b + 64 * 64;
  u16b* bfB = bfA + (size_t)N * 256;
  u16b* bfC = bfB + (size_t)N * 256;
  u16b* x_bf     = bfA;
  u16b* conv1_bf = bfA;                          // after x_bf dies (post G1)
  u16b* emb0_bf  = bfA + (size_t)N * H1;
  u16b* h1_bf    = bfB;
  u16b* prev0_bf = bfB;                          // after h1_bf dies (post G2)
  u16b* prev1_bf = bfB + (size_t)N * H1;
  u16b* h2_bf    = bfC;
  u16b* conv2_bf = bfC;                          // after h2_bf dies (post G3)

  float* pred = (float*)d_out;                   // [L]
  float* emb0 = pred + L;                        // [N, H1]
  float* emb1 = emb0 + (size_t)N * H1;           // [N, H2]

  const int* src = ei;
  const int* dst = ei + E;
  const dim3 gblk(128);

  // ---- degrees / symmetric norm ----
  k_fill<<<cdiv(N, 256), 256, 0, stream>>>(deg, N, 0.0f);
  k_deg <<<cdiv(E, 256), 256, 0, stream>>>(dst, deg, E);
  k_dinv<<<cdiv(N, 256), 256, 0, stream>>>(deg, dinv, N);

  // ---- one-time bf16 weight transposes + x conversion ----
  k_wt<<<cdiv(256 * 256, 256), 256, 0, stream>>>(Wpre1, WTpre1, 256, 256);
  k_wt<<<cdiv(128 * 256, 256), 256, 0, stream>>>(Wpre2, WTpre2, 256, 128);
  k_wt<<<cdiv(128 * 128, 256), 256, 0, stream>>>(Wc1, WTc1, 128, 128);
  k_wt<<<cdiv(128 * 128, 256), 256, 0, stream>>>(Wm1, WTm1a, 128, 128);
  k_wt<<<cdiv(128 * 128, 256), 256, 0, stream>>>(Wm1 + (size_t)H1 * H1, WTm1b, 128, 128);
  k_wt<<<cdiv(64 * 128, 256), 256, 0, stream>>>(Wc2, WTc2, 128, 64);
  k_wt<<<cdiv(64 * 64, 256), 256, 0, stream>>>(Wm2, WTm2a, 64, 64);
  k_wt<<<cdiv(64 * 64, 256), 256, 0, stream>>>(Wm2 + (size_t)H2 * H2, WTm2b, 64, 64);
  k_f2b<<<cdiv((long long)N * DIN / 2, 256), 256, 0, stream>>>(x, x_bf, (long long)N * DIN / 2);

  // ---- pre-MLP: h1 = relu(x@Wpre1+b) (bf16 only); h2 = relu(h1@Wpre2+b) ----
  k_gemm_bf<<<dim3(cdiv(N, BM), DIN / BN), gblk, 0, stream>>>(
      x_bf, DIN, WTpre1, DIN, bpre1, nullptr, h1_bf, DIN, N, DIN, DIN, 1, 0);
  k_gemm_bf<<<dim3(cdiv(N, BM), H1 / BN), gblk, 0, stream>>>(
      h1_bf, DIN, WTpre2, DIN, bpre2, nullptr, h2_bf, H1, N, H1, DIN, 1, 0);

  // prev0/prev1 bf16 copies (bfB region is free once G2 retires)
  k_f2b<<<cdiv((long long)N * H1 / 2, 256), 256, 0, stream>>>(prev0, prev0_bf, (long long)N * H1 / 2);
  k_f2b<<<cdiv((long long)N * H2 / 2, 256), 256, 0, stream>>>(prev1, prev1_bf, (long long)N * H2 / 2);

  // ---- GCN conv 1 ----
  k_gemm_bf<<<dim3(cdiv(N, BM), H1 / BN), gblk, 0, stream>>>(
      h2_bf, H1, WTc1, H1, nullptr, hw1, nullptr, H1, N, H1, H1, 0, 0);
  k_fill<<<cdiv((long long)N * H1, 256), 256, 0, stream>>>(agg1, (long long)N * H1, 0.0f);
  k_scatter<<<cdiv((long long)E * (H1 / 4), 256), 256, 0, stream>>>(
      hw1, dinv, src, dst, agg1, (long long)E * (H1 / 4), H1, 5);
  k_gcn_epilogue<<<cdiv((long long)N * H1, 256), 256, 0, stream>>>(
      agg1, hw1, dinv, bc1, conv1_bf, (long long)N * H1, H1, 7);

  // ---- temporal MLP 1: emb0 = conv1@Wm1[:H1] + prev0@Wm1[H1:] + bm1 ----
  k_gemm_bf<<<dim3(cdiv(N, BM), H1 / BN), gblk, 0, stream>>>(
      conv1_bf, H1, WTm1a, H1, bm1, emb0, nullptr, H1, N, H1, H1, 0, 0);
  k_gemm_bf<<<dim3(cdiv(N, BM), H1 / BN), gblk, 0, stream>>>(
      prev0_bf, H1, WTm1b, H1, nullptr, emb0, emb0_bf, H1, N, H1, H1, 0, 1);

  // ---- GCN conv 2 ----
  k_gemm_bf<<<dim3(cdiv(N, BM), H2 / BN), gblk, 0, stream>>>(
      emb0_bf, H1, WTc2, H1, nullptr, hw2, nullptr, H2, N, H2, H1, 0, 0);
  k_fill<<<cdiv((long long)N * H2, 256), 256, 0, stream>>>(agg2, (long long)N * H2, 0.0f);
  k_scatter<<<cdiv((long long)E * (H2 / 4), 256), 256, 0, stream>>>(
      hw2, dinv, src, dst, agg2, (long long)E * (H2 / 4), H2, 4);
  k_gcn_epilogue<<<cdiv((long long)N * H2, 256), 256, 0, stream>>>(
      agg2, hw2, dinv, bc2, conv2_bf, (long long)N * H2, H2, 6);

  // ---- temporal MLP 2: emb1 = conv2@Wm2[:H2] + prev1@Wm2[H2:] + bm2 ----
  k_gemm_bf<<<dim3(cdiv(N, BM), H2 / BN), gblk, 0, stream>>>(
      conv2_bf, H2, WTm2a, H2, bm2, emb1, nullptr, H2, N, H2, H2, 0, 0);
  k_gemm_bf<<<dim3(cdiv(N, BM), H2 / BN), gblk, 0, stream>>>(
      prev1_bf, H2, WTm2b, H2, nullptr, emb1, nullptr, H2, N, H2, H2, 0, 1);

  // ---- hadamard decoder ----
  k_decode<<<cdiv(L, 256), 256, 0, stream>>>(
      emb1, eli, eli + L, Wpost, bpost, pred, L, H2);
}
